// FullAttention_38001870635110
// MI455X (gfx1250) — matrix operand order repaired and checked by reading, not compiled, not run on hardware
//
#include <hip/hip_runtime.h>
#include <hip/hip_bf16.h>

typedef __attribute__((ext_vector_type(16))) _Float16 v16h;
typedef __attribute__((ext_vector_type(8)))  _Float16 h8;
typedef __attribute__((ext_vector_type(8)))  float    v8f;

// ---------------------------------------------------------------------------
// Async DMA: 16 bytes global -> LDS, tracked by ASYNCcnt (no VGPR round-trip).
// GV addressing: %0 = LDS byte address (vgpr), %1 = 64-bit global addr (vgpr pair).
// ---------------------------------------------------------------------------
__device__ __forceinline__ void async_cp16(void* lds_dst, const void* gsrc) {
  asm volatile("global_load_async_to_lds_b128 %0, %1, off"
               :: "v"((unsigned int)(uintptr_t)lds_dst),
                  "v"((unsigned long long)(uintptr_t)gsrc)
               : "memory");
}
__device__ __forceinline__ void wait_async0() {
  asm volatile("s_wait_asynccnt 0" ::: "memory");
}

// ---------------------------------------------------------------------------
// Load a 16x32 f16 A/B fragment (NT layout: each of 16 rows is K-contiguous).
// Per ISA layout: lane (r, half) holds row r, elements K(e) = (e/8)*16 + half*8 + e%8.
// rowbase must point at the 32-half K-chunk of this lane's row; 16B aligned.
// ---------------------------------------------------------------------------
__device__ __forceinline__ v16h frag_ld(const _Float16* rowbase, int halfsel) {
  const h8* p0 = (const h8*)(rowbase + halfsel * 8);
  const h8* p1 = (const h8*)(rowbase + 16 + halfsel * 8);
  h8 a = *p0, b = *p1;
  v16h f;
#pragma unroll
  for (int i = 0; i < 8; ++i) { f[i] = a[i]; f[8 + i] = b[i]; }
  return f;
}

// ---------------------------------------------------------------------------
// f32 -> f16 conversion (vectorized x4)
// ---------------------------------------------------------------------------
__global__ void f32_to_f16_kernel(const float* __restrict__ in,
                                  _Float16* __restrict__ out, int n) {
  int i = (blockIdx.x * blockDim.x + threadIdx.x) * 4;
  if (i < n) {
    float4 v = *(const float4*)(in + i);
    out[i + 0] = (_Float16)v.x;
    out[i + 1] = (_Float16)v.y;
    out[i + 2] = (_Float16)v.z;
    out[i + 3] = (_Float16)v.w;
  }
}

// ---------------------------------------------------------------------------
// NT GEMM: C[M,N] = A[M,K] * B[N,K]^T + bias[N]
// MODE 0: QKV epilogue -> scatter f16 Q(scaled by 0.125)/K/V into [B,H,T,64]
// MODE 1: f32 output + bias straight to d_out
// Block tile 128x128, BK=32, 8 waves; each wave computes 32x64.
// Double-buffered async global->LDS staging: DMA of tile i+1 overlaps WMMA on i.
// ---------------------------------------------------------------------------
template <int MODE>
__global__ __launch_bounds__(256) void gemm_nt_kernel(
    const _Float16* __restrict__ A, const _Float16* __restrict__ Bm,
    const float* __restrict__ bias, float* __restrict__ outF,
    _Float16* __restrict__ Qo, _Float16* __restrict__ Ko,
    _Float16* __restrict__ Vo, int M, int N, int K) {
  __shared__ _Float16 As[2][128 * 40];  // stride 40 halfs = 80B, bank-conflict-free
  __shared__ _Float16 Bs[2][128 * 40];

  const int tid = threadIdx.x;
  const int w = tid >> 5, lane = tid & 31;
  const int r = lane & 15, halfsel = lane >> 4;
  const int wm = w & 3;   // wave row: 4 x 32 rows
  const int wn = w >> 2;  // wave col: 2 x 64 cols
  const int m0 = blockIdx.x * 128, n0 = blockIdx.y * 128;

  // per-thread staging slot: row tr, 16-half chunk tc
  const int tr = tid >> 1, tc = tid & 1;
  const _Float16* agp = A + (size_t)(m0 + tr) * K + tc * 16;
  const _Float16* bgp = Bm + (size_t)(n0 + tr) * K + tc * 16;
  const int lslot = tr * 40 + tc * 16;

  v8f acc[2][4] = {};

  // prologue: DMA tile 0 into buffer 0
  async_cp16(&As[0][lslot], agp);
  async_cp16(&As[0][lslot + 8], agp + 8);
  async_cp16(&Bs[0][lslot], bgp);
  async_cp16(&Bs[0][lslot + 8], bgp + 8);

  const int nsteps = K / 32;
  for (int i = 0; i < nsteps; ++i) {
    wait_async0();      // my DMA for buf[i&1] has landed
    __syncthreads();    // everyone's has; prev iteration's reads of buf[(i+1)&1] done
    if (i + 1 < nsteps) {
      async_cp16(&As[(i + 1) & 1][lslot], agp + (i + 1) * 32);
      async_cp16(&As[(i + 1) & 1][lslot + 8], agp + (i + 1) * 32 + 8);
      async_cp16(&Bs[(i + 1) & 1][lslot], bgp + (i + 1) * 32);
      async_cp16(&Bs[(i + 1) & 1][lslot + 8], bgp + (i + 1) * 32 + 8);
    }
    const _Float16* Ab = As[i & 1];
    const _Float16* Bb = Bs[i & 1];

    v16h af[2], bf[4];
#pragma unroll
    for (int mi = 0; mi < 2; ++mi)
      af[mi] = frag_ld(Ab + (wm * 32 + mi * 16 + r) * 40, halfsel);
#pragma unroll
    for (int ni = 0; ni < 4; ++ni)
      bf[ni] = frag_ld(Bb + (wn * 64 + ni * 16 + r) * 40, halfsel);
#pragma unroll
    for (int mi = 0; mi < 2; ++mi)
#pragma unroll
      for (int ni = 0; ni < 4; ++ni)
        acc[mi][ni] = __builtin_amdgcn_wmma_f32_16x16x32_f16(
            false, af[mi], false, bf[ni], (short)0, acc[mi][ni], false, false);
  }

  // Epilogue. D layout: element j at lane(r,half) = (M = j + 8*half, N = r).
#pragma unroll
  for (int mi = 0; mi < 2; ++mi)
#pragma unroll
    for (int ni = 0; ni < 4; ++ni)
#pragma unroll
      for (int j = 0; j < 8; ++j) {
        int gm = m0 + wm * 32 + mi * 16 + halfsel * 8 + j;
        int gn = n0 + wn * 64 + ni * 16 + r;
        float v = acc[mi][ni][j] + bias[gn];
        if (MODE == 0) {
          int which = gn >> 10, c = gn & 1023;
          int h = c >> 6, d = c & 63;
          int b = gm >> 11, t = gm & 2047;  // M = 2*2048
          size_t idx = ((size_t)((b * 16 + h) * 2048 + t)) * 64 + d;
          if (which == 0)      Qo[idx] = (_Float16)(v * 0.125f);  // fold 1/sqrt(64)
          else if (which == 1) Ko[idx] = (_Float16)v;
          else                 Vo[idx] = (_Float16)v;
        } else {
          outF[(size_t)gm * N + gn] = v;
        }
      }
}

// ---------------------------------------------------------------------------
// Causal flash attention. Grid: (T/128, B*H). 8 waves; wave w owns 16 q-rows.
// Q pre-scaled by 1/sqrt(HD). Output written f16 as [B,T,H*64] for out-proj.
// K block staged by async DMA; V transposed through registers (needs 16-bit
// scatter, below async granularity).
// ---------------------------------------------------------------------------
__global__ __launch_bounds__(256) void attn_kernel(
    const _Float16* __restrict__ Qg, const _Float16* __restrict__ Kg,
    const _Float16* __restrict__ Vg, _Float16* __restrict__ Og) {
  __shared__ _Float16 Ksh[64 * 72];      // K block, row-major [key][d]
  __shared__ _Float16 Vt[64 * 72];       // V block transposed [d][key]
  __shared__ _Float16 Ps[8 * 16 * 72];   // per-wave P staging [16 rows][64 keys]

  const int tid = threadIdx.x;
  const int w = tid >> 5, lane = tid & 31;
  const int r = lane & 15, halfsel = lane >> 4;
  const int bh = blockIdx.y, qb = blockIdx.x;
  const int q0 = qb * 128;

  const _Float16* Qb = Qg + (size_t)bh * 2048 * 64;
  const _Float16* Kb = Kg + (size_t)bh * 2048 * 64;
  const _Float16* Vb = Vg + (size_t)bh * 2048 * 64;

  // Q fragments for this wave's 16 rows, K-dim = 64 -> 2 chunks (register-resident)
  v16h qf[2];
  {
    const _Float16* qrow = Qb + (size_t)(q0 + w * 16 + r) * 64;
    qf[0] = frag_ld(qrow, halfsel);
    qf[1] = frag_ld(qrow + 32, halfsel);
  }

  v8f oacc[4] = {};
  float mrow[8], lrow[8];
#pragma unroll
  for (int j = 0; j < 8; ++j) { mrow[j] = -1e30f; lrow[j] = 0.f; }

  const int kbmax = 2 * qb + 2;  // causal: keys [0, 128*(qb+1))
  for (int kb = 0; kb < kbmax; ++kb) {
    __syncthreads();  // previous iteration's readers are done
    {  // stage K (async DMA) and V transposed (register path)
      int tr = tid >> 2, c = tid & 3;
      const _Float16* kg = Kb + (size_t)(kb * 64 + tr) * 64 + c * 16;
      async_cp16(Ksh + tr * 72 + c * 16, kg);
      async_cp16(Ksh + tr * 72 + c * 16 + 8, kg + 8);
      const h8* sv = (const h8*)(Vb + (size_t)(kb * 64 + tr) * 64 + c * 16);
      h8 v0 = sv[0], v1 = sv[1];
#pragma unroll
      for (int i = 0; i < 8; ++i) {
        Vt[(c * 16 + i) * 72 + tr]     = v0[i];
        Vt[(c * 16 + 8 + i) * 72 + tr] = v1[i];
      }
    }
    wait_async0();
    __syncthreads();

    // wave-uniform skip of fully-masked key blocks
    bool skip = (kb * 64) > (q0 + w * 16 + 15);
    if (!skip) {
      // S = Q * K^T  (16 x 64)
      v8f s[4] = {};
#pragma unroll
      for (int ni = 0; ni < 4; ++ni)
#pragma unroll
        for (int kc = 0; kc < 2; ++kc) {
          v16h bf = frag_ld(Ksh + (ni * 16 + r) * 72 + kc * 32, halfsel);
          s[ni] = __builtin_amdgcn_wmma_f32_16x16x32_f16(
              false, qf[kc], false, bf, (short)0, s[ni], false, false);
        }

      if (kb >= 2 * qb) {  // diagonal blocks: apply causal mask
#pragma unroll
        for (int ni = 0; ni < 4; ++ni)
#pragma unroll
          for (int j = 0; j < 8; ++j) {
            int kk = kb * 64 + ni * 16 + r;
            int qq = q0 + w * 16 + halfsel * 8 + j;
            if (kk > qq) s[ni][j] = -1e30f;
          }
      }

      // online softmax: rows live in 16-lane half-groups -> shfl_xor reduce
#pragma unroll
      for (int j = 0; j < 8; ++j) {
        float mx = fmaxf(fmaxf(s[0][j], s[1][j]), fmaxf(s[2][j], s[3][j]));
#pragma unroll
        for (int off = 1; off < 16; off <<= 1)
          mx = fmaxf(mx, __shfl_xor(mx, off, 32));
        float nm = fmaxf(mrow[j], mx);
        float corr = __expf(mrow[j] - nm);
        mrow[j] = nm;
        float rs = 0.f;
#pragma unroll
        for (int ni = 0; ni < 4; ++ni) {
          float p = __expf(s[ni][j] - nm);
          s[ni][j] = p;
          rs += p;
        }
#pragma unroll
        for (int off = 1; off < 16; off <<= 1)
          rs += __shfl_xor(rs, off, 32);
        lrow[j] = lrow[j] * corr + rs;
#pragma unroll
        for (int di = 0; di < 4; ++di) oacc[di][j] *= corr;
      }

      // round-trip P through per-wave LDS to get A-fragment layout
      _Float16* Pw = Ps + w * 16 * 72;
#pragma unroll
      for (int ni = 0; ni < 4; ++ni)
#pragma unroll
        for (int j = 0; j < 8; ++j)
          Pw[(halfsel * 8 + j) * 72 + ni * 16 + r] = (_Float16)s[ni][j];
      // same-wave LDS RAW: DS ops are in-order per wave; just drain dscnt
      asm volatile("s_wait_dscnt 0" ::: "memory");

      // O += P * V   (K-dim = 64 keys -> 2 chunks)
#pragma unroll
      for (int di = 0; di < 4; ++di)
#pragma unroll
        for (int kc = 0; kc < 2; ++kc) {
          v16h pf = frag_ld(Pw + r * 72 + kc * 32, halfsel);
          v16h vf = frag_ld(Vt + (di * 16 + r) * 72 + kc * 32, halfsel);
          oacc[di] = __builtin_amdgcn_wmma_f32_16x16x32_f16(
              false, pf, false, vf, (short)0, oacc[di], false, false);
        }
    }
  }

  // epilogue: normalize and scatter f16 into [B, T, H*64]
  const int b = bh >> 4, h = bh & 15;
#pragma unroll
  for (int j = 0; j < 8; ++j) {
    float inv = 1.0f / lrow[j];
    int tq = q0 + w * 16 + halfsel * 8 + j;
    size_t rowbase = ((size_t)(b * 2048 + tq)) * 1024 + h * 64;
#pragma unroll
    for (int di = 0; di < 4; ++di)
      Og[rowbase + di * 16 + r] = (_Float16)(oacc[di][j] * inv);
  }
}

// ---------------------------------------------------------------------------
extern "C" void kernel_launch(void* const* d_in, const int* in_sizes, int n_in,
                              void* d_out, int out_size, void* d_ws,
                              size_t ws_size, hipStream_t stream) {
  const float* x     = (const float*)d_in[0];
  // d_in[1]: causal mask (computed analytically, unused)
  const float* qkv_w = (const float*)d_in[2];
  const float* qkv_b = (const float*)d_in[3];
  const float* out_w = (const float*)d_in[4];
  const float* out_b = (const float*)d_in[5];
  float* out = (float*)d_out;

  const size_t N_X    = (size_t)4096 * 1024;  // B*T x C
  const size_t N_WQKV = (size_t)3072 * 1024;
  const size_t N_WOUT = (size_t)1024 * 1024;
  const size_t N_HEAD = (size_t)2 * 16 * 2048 * 64;  // per Q/K/V tensor

  _Float16* wsh   = (_Float16*)d_ws;
  _Float16* xh    = wsh;
  _Float16* wqkvh = xh + N_X;
  _Float16* wouth = wqkvh + N_WQKV;
  _Float16* Qh    = wouth + N_WOUT;
  _Float16* Kh    = Qh + N_HEAD;
  _Float16* Vh    = Kh + N_HEAD;
  _Float16* Ah    = xh;  // x is dead after QKV GEMM; reuse for attention output

  auto cvt = [&](const float* s, _Float16* d, int n) {
    f32_to_f16_kernel<<<dim3((n / 4 + 255) / 256), 256, 0, stream>>>(s, d, n);
  };
  cvt(x, xh, (int)N_X);
  cvt(qkv_w, wqkvh, (int)N_WQKV);
  cvt(out_w, wouth, (int)N_WOUT);

  // QKV projection: [4096,1024] x [3072,1024]^T
  gemm_nt_kernel<0><<<dim3(32, 24), 256, 0, stream>>>(
      xh, wqkvh, qkv_b, nullptr, Qh, Kh, Vh, 4096, 3072, 1024);

  // causal flash attention over 32 (b,h) pairs, 16 q-blocks each
  attn_kernel<<<dim3(16, 32), 256, 0, stream>>>(Qh, Kh, Vh, Ah);

  // output projection: [4096,1024] x [1024,1024]^T -> f32 d_out
  gemm_nt_kernel<1><<<dim3(32, 8), 256, 0, stream>>>(
      Ah, wouth, out_b, out, nullptr, nullptr, nullptr, 4096, 1024, 1024);
}
